// GCMCLayer_39041252721056
// MI455X (gfx1250) — compile-verified, matrix operand order checked by reading, not run on hardware
//
#include <hip/hip_runtime.h>
#include <hip/hip_bf16.h>

// ---------------------------------------------------------------------------
// GCMC layer for MI455X (gfx1250, wave32, WMMA, TDM).
//   Phase 1: W[r] = sum_b att[r,b] * basis[b];  fc_w -> bf16 swizzled into
//            WMMA B-fragment layout (32B contiguous per lane per fragment)
//   Phase 2: fused embed + edge segment-sum (global_atomic_add_f32), W[r]
//            (256KB) staged into LDS via tensor_load_to_lds (TDM)
//   Phase 3: out[100000,256] = bf16(H*ci) @ bf16(fc_w) + fc_b
//            via v_wmma_f32_16x16x32_bf16, 8 WMMA / wave / K-step
// ---------------------------------------------------------------------------

typedef __attribute__((ext_vector_type(16))) __bf16   v16bf;
typedef __attribute__((ext_vector_type(8)))  float    v8f;
typedef __attribute__((ext_vector_type(4)))  uint32_t u32x4;
typedef __attribute__((ext_vector_type(8)))  uint32_t u32x8;

#define N_NODES   50000
#define N_EDGES   500000
#define R_RATINGS 5
#define IN_UNITS  1024
#define MU        64
#define EMB       192            // 3*MU
#define KDIM      960            // R_RATINGS*EMB
#define OUT_UNITS 256
#define M_TOTAL   (2 * N_NODES)  // 100000 rows (drug then dis); 32 | M_TOTAL

#define W_ELEMS   (R_RATINGS * IN_UNITS * MU)   // 327680 floats
#define WR_ELEMS  (IN_UNITS * MU)               // 65536 floats = 256KB

// ---------------------------------------------------------------------------
// Kernel 1: W[r][i][m] = sum_b att[r][b] * basis[b][i][m]
// ---------------------------------------------------------------------------
__global__ void build_w_kernel(const float* __restrict__ att,
                               const float* __restrict__ basis,
                               float* __restrict__ W) {
    int idx = blockIdx.x * blockDim.x + threadIdx.x;
    if (idx >= W_ELEMS) return;
    int r  = idx / WR_ELEMS;
    int im = idx - r * WR_ELEMS;
    float acc = 0.f;
#pragma unroll
    for (int b = 0; b < 4; ++b)
        acc += att[r * 4 + b] * basis[b * WR_ELEMS + im];
    W[idx] = acc;
}

// ---------------------------------------------------------------------------
// Kernel 2: fc_w fp32 [960,256] -> bf16 swizzled into WMMA B-fragment layout.
//   Wbs[kt][nt][lane][j], j=0..15:  (16-bit B 32x16, mirror of A layout)
//     N = nt*16 + (lane&15)
//     K = kt*32 + ((lane>>4)<<3) + (j&7) + ((j>>3)<<4)
//   => each (kt,nt,lane) fragment is 32 contiguous bytes.
// ---------------------------------------------------------------------------
__global__ void cvt_fcw_swz_kernel(const float* __restrict__ fcw,
                                   __bf16* __restrict__ wbs) {
    int idx = blockIdx.x * blockDim.x + threadIdx.x;
    if (idx >= KDIM * OUT_UNITS) return;
    int j    = idx & 15;
    int lane = (idx >> 4) & 31;
    int nt   = (idx >> 9) & 15;
    int kt   = idx >> 13;                 // 0..29
    int K = kt * 32 + ((lane >> 4) << 3) + (j & 7) + ((j >> 3) << 4);
    int N = nt * 16 + (lane & 15);
    wbs[idx] = (__bf16)fcw[K * OUT_UNITS + N];
}

// ---------------------------------------------------------------------------
// Kernel 3: fused embed + scatter (segment sum).
//   grid.y = r*2 + d.   d==0: drug -> dis (H rows [50000,100000))
//                       d==1: dis  -> drug (H rows [0,50000))
//   W[r] (256KB) pulled into LDS by the Tensor Data Mover (2D tile 16x4096
//   f32, contiguous), then each wave scatters 192 cols/edge via f32 atomics.
// ---------------------------------------------------------------------------
#define SCAT_THREADS    256
#define EDGES_PER_BLOCK 2048

__global__ void __launch_bounds__(SCAT_THREADS)
scatter_kernel(const int*   __restrict__ drug_feat,
               const int*   __restrict__ dis_feat,
               const int*   __restrict__ src,
               const int*   __restrict__ dst,
               const float* __restrict__ cj_drug,
               const float* __restrict__ cj_dis,
               const float* __restrict__ W,
               float*       __restrict__ H) {
    extern __shared__ float lds_w[];   // WR_ELEMS floats = 256KB

    const int rd = blockIdx.y;
    const int r  = rd >> 1;
    const int d  = rd & 1;

    // --- TDM: DMA W[r] (65536 f32 as a 16 x 4096 tile) into LDS -----------
    if (threadIdx.x < 32) {
        const uint64_t ga   = (uint64_t)(uintptr_t)(W + (size_t)r * WR_ELEMS);
        const uint32_t ldsa = (uint32_t)(uintptr_t)(void*)lds_w;
        // D# group0: count=1 | lds_addr | global_addr[56:0] | type=2
        u32x4 g0 = { 1u,
                     ldsa,
                     (uint32_t)ga,
                     (uint32_t)((ga >> 32) & 0x01FFFFFFu) | 0x80000000u };
        // D# group1: wg_mask=0, data_size=2 (4B); tensor_dim0=4096,
        // tensor_dim1=16, tile_dim0=4096, tile_dim1=16, tile_dim2=0,
        // tensor_dim0_stride=4096, tensor_dim1_stride=0
        u32x8 g1 = { 0x00020000u,   // [17:16] data_size=2
                     0x10000000u,   // tensor_dim0[15:0]=4096 @ [31:16]
                     0x00100000u,   // tensor_dim0[31:16]=0, tensor_dim1[15:0]=16
                     0x10000000u,   // tensor_dim1[31:16]=0, tile_dim0=4096
                     0x00000010u,   // tile_dim1=16, tile_dim2=0
                     4096u,         // tensor_dim0_stride[31:0]
                     0u, 0u };
        u32x4 gz = { 0u, 0u, 0u, 0u };   // groups 2/3 unused (2D tensor)
        asm volatile("tensor_load_to_lds %0, %1, %2, %3"
                     :: "s"(g0), "s"(g1), "s"(gz), "s"(gz)
                     : "memory");
#if __has_builtin(__builtin_amdgcn_s_wait_tensorcnt)
        __builtin_amdgcn_s_wait_tensorcnt(0);
#else
        asm volatile("s_wait_tensorcnt 0x0" ::: "memory");
#endif
    }
    __syncthreads();

    const int* sn   = (d == 0) ? (src + r * N_EDGES) : (dst + r * N_EDGES);
    const int* dn   = (d == 0) ? (dst + r * N_EDGES) : (src + r * N_EDGES);
    const int* feat = (d == 0) ? drug_feat : dis_feat;
    const float* cj = (d == 0) ? cj_drug : cj_dis;
    const int rowoff = (d == 0) ? N_NODES : 0;

    const int lane = threadIdx.x & 31;
    const int wave = threadIdx.x >> 5;
    const int nwaves = SCAT_THREADS / 32;

    int e0   = blockIdx.x * EDGES_PER_BLOCK;
    int eEnd = e0 + EDGES_PER_BLOCK;
    if (eEnd > N_EDGES) eEnd = N_EDGES;

    for (int e = e0 + wave; e < eEnd; e += nwaves) {
        const int s = sn[e];
        const int t = dn[e];
        const int f0 = feat[s * 3 + 0];
        const int f1 = feat[s * 3 + 1];
        const int f2 = feat[s * 3 + 2];
        const float c = cj[s];
        float* hrow = H + (size_t)(rowoff + t) * KDIM + r * EMB;
#pragma unroll
        for (int j = 0; j < 6; ++j) {
            const int col = lane + 32 * j;        // 0..191
            const int k   = col >> 6;             // which of the 3 gathers
            const int cc  = col & 63;
            const int fi  = (k == 0) ? f0 : ((k == 1) ? f1 : f2);
            const float v = lds_w[fi * MU + cc] * c;
            atomicAdd(&hrow[col], v);             // global_atomic_add_f32
        }
    }
}

// ---------------------------------------------------------------------------
// Kernel 4: out[M_TOTAL,256] = bf16(H*ci) @ fc_w + fc_b
//   256 threads = 8 waves; block tile 64 rows x 256 cols.
//   Wave (wy,wx): 32 rows x 64 cols = 2 A-frags x 4 B-frags = 8 WMMAs/K-step.
//   B fragments: single 32B coalesced load from the pre-swizzled Wbs (L2).
//   M_TOTAL % 32 == 0, so each 32-row wave tile is fully valid or fully
//   out of range -> uniform early-exit, unconditional loads/stores.
// ---------------------------------------------------------------------------
#define GEMM_THREADS 256

__global__ void __launch_bounds__(GEMM_THREADS)
gemm_kernel(const float*  __restrict__ H,
            const __bf16* __restrict__ Wbs,
            const float*  __restrict__ fc_b,
            const float*  __restrict__ ci_drug,
            const float*  __restrict__ ci_dis,
            float*        __restrict__ out) {
    const int lane = threadIdx.x & 31;
    const int wave = threadIdx.x >> 5;      // 0..7
    const int wy = wave >> 2;               // 0..1
    const int wx = wave & 3;                // 0..3

    const int rowbase = blockIdx.x * 64 + wy * 32;
    if (rowbase >= M_TOTAL) return;         // wave-uniform tail cut (no barriers)

    const int col0   = wx * 64;
    const int ntile0 = col0 >> 4;           // first of 4 N-tiles

    // A-fragment lane mapping (16-bit A 16x32, ISA 7.12.2):
    //   lanes 0-15:  M=lane,    K in {0..7} u {16..23}
    //   lanes 16-31: M=lane-16, K in {8..15} u {24..31}
    const int m    = lane & 15;
    const int koff = (lane >> 4) << 3;      // 0 or 8

    float ci[2];
    const float* arow[2];
#pragma unroll
    for (int i = 0; i < 2; ++i) {
        const int row = rowbase + i * 16 + m;          // always < M_TOTAL
        ci[i]   = (row < N_NODES) ? ci_drug[row] : ci_dis[row - N_NODES];
        arow[i] = H + (size_t)row * KDIM;
    }

    v8f acc[2][4] = {};

    for (int kb = 0; kb < KDIM; kb += 32) {
        v16bf a[2];
#pragma unroll
        for (int i = 0; i < 2; ++i) {
            const float* p0 = arow[i] + kb + koff;
            const float* p1 = arow[i] + kb + 16 + koff;
#pragma unroll
            for (int j = 0; j < 8; ++j) {
                a[i][j]     = (__bf16)(p0[j] * ci[i]);
                a[i][j + 8] = (__bf16)(p1[j] * ci[i]);
            }
        }
        const int ktile = kb >> 5;
#pragma unroll
        for (int t = 0; t < 4; ++t) {
            const v16bf b = *(const v16bf*)(
                Wbs + (((size_t)(ktile * 16 + ntile0 + t) * 32 + lane) << 4));
            acc[0][t] = __builtin_amdgcn_wmma_f32_16x16x32_bf16(
                false, a[0], false, b, (short)0, acc[0][t], false, false);
            acc[1][t] = __builtin_amdgcn_wmma_f32_16x16x32_bf16(
                false, a[1], false, b, (short)0, acc[1][t], false, false);
        }
    }

    // C/D layout (32-bit 16x16): lane holds N = lane&15; M = (lane>>4)*8 + g.
    // Output is write-once/never-read: stream it (non-temporal).
    const int mbase = (lane >> 4) * 8;
#pragma unroll
    for (int t = 0; t < 4; ++t) {
        const int n = col0 + t * 16 + m;
        const float bias = fc_b[n];
#pragma unroll
        for (int i = 0; i < 2; ++i) {
            float* orow = out + (size_t)(rowbase + i * 16 + mbase) * OUT_UNITS + n;
#pragma unroll
            for (int g = 0; g < 8; ++g) {
                __builtin_nontemporal_store(acc[i][t][g] + bias,
                                            orow + (size_t)g * OUT_UNITS);
            }
        }
    }
}

// ---------------------------------------------------------------------------
// Launch
// ---------------------------------------------------------------------------
extern "C" void kernel_launch(void* const* d_in, const int* in_sizes, int n_in,
                              void* d_out, int out_size, void* d_ws, size_t ws_size,
                              hipStream_t stream) {
    const int*   drug_feat = (const int*)d_in[0];
    const int*   dis_feat  = (const int*)d_in[1];
    const int*   src       = (const int*)d_in[2];
    const int*   dst       = (const int*)d_in[3];
    const float* cj_drug   = (const float*)d_in[4];
    const float* ci_drug   = (const float*)d_in[5];
    const float* cj_dis    = (const float*)d_in[6];
    const float* ci_dis    = (const float*)d_in[7];
    const float* att       = (const float*)d_in[8];
    const float* basis     = (const float*)d_in[9];
    const float* fc_w      = (const float*)d_in[10];
    const float* fc_b      = (const float*)d_in[11];
    float* out = (float*)d_out;

    // Workspace layout (32B aligned):
    //   W   : fp32 [5][1024][64]             = 1,310,720 B
    //   Wbs : bf16 swizzled [30][16][32][16] =   491,520 B
    //   H   : fp32 [100000][960]             = 384,000,000 B
    char* ws = (char*)d_ws;
    float*  W   = (float*)ws;
    __bf16* Wbs = (__bf16*)(ws + (size_t)W_ELEMS * sizeof(float));
    float*  H   = (float*)(ws + (size_t)W_ELEMS * sizeof(float)
                              + (size_t)KDIM * OUT_UNITS * sizeof(__bf16));

    // H must be zero before atomic accumulation (deterministic per call).
    hipMemsetAsync(H, 0, (size_t)M_TOTAL * KDIM * sizeof(float), stream);

    build_w_kernel<<<(W_ELEMS + 255) / 256, 256, 0, stream>>>(att, basis, W);
    cvt_fcw_swz_kernel<<<(KDIM * OUT_UNITS + 255) / 256, 256, 0, stream>>>(fc_w, Wbs);

    dim3 sgrid((N_EDGES + EDGES_PER_BLOCK - 1) / EDGES_PER_BLOCK, 2 * R_RATINGS);
    scatter_kernel<<<sgrid, SCAT_THREADS, WR_ELEMS * sizeof(float), stream>>>(
        drug_feat, dis_feat, src, dst, cj_drug, cj_dis, W, H);

    gemm_kernel<<<(M_TOTAL + 63) / 64, GEMM_THREADS, 0, stream>>>(
        H, Wbs, fc_b, ci_drug, ci_dis, out);
}
